// ICMRecurrent_42898133353462
// MI455X (gfx1250) — matrix operand order repaired and checked by reading, not compiled, not run on hardware
//
#include <hip/hip_runtime.h>
#include <math.h>

// ---------------- problem constants ----------------
constexpr int T1 = 257;          // states timesteps (T+1)
constexpr int T  = 256;          // action timesteps
constexpr int B  = 512;
constexpr int SD = 128;
constexpr int AD = 32;
constexpr int HS = 512;
constexpr int SF = 256;

// ---------------- WMMA plumbing ----------------
typedef __attribute__((ext_vector_type(16))) __bf16        v16bf;
typedef __attribute__((ext_vector_type(8)))  float         v8f;
typedef __attribute__((ext_vector_type(4)))  unsigned int  v4u;
typedef __attribute__((ext_vector_type(4)))  int           i4;

union BFrag {
  v16bf          bf;
  v4u            q[2];
  unsigned short us[16];
};

static __device__ inline unsigned short f2bf(float x) {
  union { float f; unsigned u; } v; v.f = x;
  unsigned r = v.u + 0x7FFFu + ((v.u >> 16) & 1u);
  return (unsigned short)(r >> 16);
}

static __device__ inline void zero_frag(BFrag& f) {
  v4u z = {0u, 0u, 0u, 0u};
  f.q[0] = z; f.q[1] = z;
}

// A fragment (16x32 bf16, row-major source): lane r=lane&15 -> row r,
// hi=lane>>4 selects K sub-blocks [hi*8..hi*8+7] and [16+hi*8..23+hi*8]
static __device__ inline void load_fragA_bf16(BFrag& f, const unsigned short* src,
                                              int ld, int r0, int k0, int lane) {
  const int r = lane & 15, hi = (lane >> 4) & 1;
  const unsigned short* p = src + (size_t)(r0 + r) * ld + k0 + hi * 8;
  f.q[0] = *(const v4u*)(p);
  f.q[1] = *(const v4u*)(p + 16);
}

// Same, but source is f32 (convert to bf16 on the fly)
static __device__ inline void load_fragA_f32(BFrag& f, const float* src,
                                             int ld, int r0, int k0, int lane) {
  const int r = lane & 15, hi = (lane >> 4) & 1;
  const float* p0 = src + (size_t)(r0 + r) * ld + k0 + hi * 8;
  const float* p1 = p0 + 16;
  float4 a = *(const float4*)(p0);
  float4 b = *(const float4*)(p0 + 4);
  float4 c = *(const float4*)(p1);
  float4 d = *(const float4*)(p1 + 4);
  f.us[0]=f2bf(a.x); f.us[1]=f2bf(a.y); f.us[2]=f2bf(a.z); f.us[3]=f2bf(a.w);
  f.us[4]=f2bf(b.x); f.us[5]=f2bf(b.y); f.us[6]=f2bf(b.z); f.us[7]=f2bf(b.w);
  f.us[8]=f2bf(c.x); f.us[9]=f2bf(c.y); f.us[10]=f2bf(c.z); f.us[11]=f2bf(c.w);
  f.us[12]=f2bf(d.x); f.us[13]=f2bf(d.y); f.us[14]=f2bf(d.z); f.us[15]=f2bf(d.w);
}

// B fragment (32x16 bf16, B = W^T, W row-major (N rows, K cols)):
// lane n=lane&15 -> column n, hi selects K sub-blocks [hi*8..hi*8+7] and [16+hi*8..23+hi*8]
static __device__ inline void load_fragB(BFrag& f, const unsigned short* W,
                                         int ld, int n0, int k0, int lane) {
  const int n = lane & 15, hi = (lane >> 4) & 1;
  const unsigned short* p = W + (size_t)(n0 + n) * ld + k0 + hi * 8;
  f.q[0] = *(const v4u*)(p);
  f.q[1] = *(const v4u*)(p + 16);
}

#define WMMA_BF16(A, Bf, C) \
  __builtin_amdgcn_wmma_f32_16x16x32_bf16(false, (A).bf, false, (Bf).bf, (short)0, (C), false, false)

static __device__ inline float sigmoidf_(float x) { return 1.f / (1.f + expf(-x)); }

// ---------------- CDNA5 async global->LDS staging ----------------
// Builtin signature (from hipcc diagnostic): param1 = int4 addrspace(1)*,
// so call as (global int4*, lds int4*, imm offset, imm cpol).
typedef __attribute__((address_space(1))) i4 gi4;
typedef __attribute__((address_space(3))) i4 li4;

static __device__ inline void async_cp16(void* lptr, const void* gptr) {
#if __has_builtin(__builtin_amdgcn_global_load_async_to_lds_b128)
  __builtin_amdgcn_global_load_async_to_lds_b128((gi4*)gptr, (li4*)lptr, 0, 0);
#else
  unsigned loff = (unsigned)(size_t)(li4*)lptr;
  unsigned long long ga = (unsigned long long)(size_t)gptr;
  asm volatile("global_load_async_to_lds_b128 %0, %1, off"
               :: "v"(loff), "v"(ga) : "memory");
#endif
}

static __device__ inline void async_wait0() {
#if __has_builtin(__builtin_amdgcn_s_wait_asynccnt)
  __builtin_amdgcn_s_wait_asynccnt(0);
#else
  asm volatile("s_wait_asynccnt 0x0" ::: "memory");
#endif
}

// ---------------- kernels ----------------

__global__ void __launch_bounds__(256)
k_f32_to_bf16(const float* __restrict__ src, unsigned short* __restrict__ dst, int n) {
  int i = blockIdx.x * blockDim.x + threadIdx.x;
  if (i < n) dst[i] = f2bf(src[i]);
}

__global__ void __launch_bounds__(256)
k_init(float* __restrict__ out, float* __restrict__ c0, float* __restrict__ c1) {
  int i = blockIdx.x * blockDim.x + threadIdx.x;
  if (i < 2) out[i] = 0.f;
  if (i < B * SF) { c0[i] = 0.f; c1[i] = 0.f; }
}

// feats = relu(states @ Wfe^T + bfe) -> bf16   M=T1*B, N=HS, K=SD
__global__ void __launch_bounds__(128)
k_feats(const float* __restrict__ X, const unsigned short* __restrict__ WfeB,
        const float* __restrict__ bfe, unsigned short* __restrict__ featsB) {
  const int lane = threadIdx.x & 31;
  const int wv   = threadIdx.x >> 5;
  const int row0 = blockIdx.x * 64 + wv * 16;
  const int col0 = blockIdx.y * 64;
  const int ll = lane & 15, hi = lane >> 4;

  const v8f vzero = {0,0,0,0,0,0,0,0};
  v8f acc[4]; BFrag a, b;
#pragma unroll
  for (int s = 0; s < 4; ++s) acc[s] = vzero;

  for (int k0 = 0; k0 < SD; k0 += 32) {
    load_fragA_f32(a, X, SD, row0, k0, lane);
#pragma unroll
    for (int s = 0; s < 4; ++s) {
      load_fragB(b, WfeB, SD, col0 + s * 16, k0, lane);
      acc[s] = WMMA_BF16(a, b, acc[s]);
    }
  }
#pragma unroll
  for (int s = 0; s < 4; ++s) {
    const int col = col0 + s * 16 + ll;
    const float bb = bfe[col];
#pragma unroll
    for (int v = 0; v < 8; ++v) {
      float x = acc[s][v] + bb;
      x = x > 0.f ? x : 0.f;
      featsB[(size_t)(row0 + hi * 8 + v) * HS + col] = f2bf(x);
    }
  }
}

// Fused 2-layer LSTM step for one timestep.
// Block: 256 threads = 8 waves; block owns 16 batch rows; wave wv owns sf
// columns [wv*32, wv*32+32) for all four gates of both layers.
// Inputs staged to LDS with async global->LDS; layer0 h handed to layer1 in LDS.
__global__ void __launch_bounds__(256)
k_lstm_step(const unsigned short* __restrict__ Xt,       // feats tile base (B, HS)
            const unsigned short* __restrict__ h0prevG,  // (B, SF) bf16
            unsigned short* __restrict__ h0curG,         // (B, SF) bf16
            float* __restrict__ h0F,                     // (B, SF) f32 (last step)
            const unsigned short* __restrict__ h1prevG,  // outsB[t-1]
            unsigned short* __restrict__ h1outB,         // outsB[t]
            float* __restrict__ h1outF,                  // outsF[t]
            const unsigned short* __restrict__ Wih0B, const unsigned short* __restrict__ Whh0B,
            const float* __restrict__ bih0, const float* __restrict__ bhh0,
            const unsigned short* __restrict__ Wih1B, const unsigned short* __restrict__ Whh1B,
            const float* __restrict__ bih1, const float* __restrict__ bhh1,
            const int* __restrict__ dones,               // B ints (this t)
            float* __restrict__ c0, float* __restrict__ c1,
            int firstStep, int lastStep) {
  __shared__ unsigned short s_x[16][512];    // 16KB feats tile
  __shared__ unsigned short s_h0p[16][256];  // 8KB  layer0 h_prev
  __shared__ unsigned short s_h1p[16][256];  // 8KB  layer1 h_prev
  __shared__ unsigned short s_h0[16][256];   // 8KB  layer0 h_new -> layer1 input

  const int tid  = threadIdx.x;
  const int lane = tid & 31;
  const int wv   = tid >> 5;        // 0..7
  const int row0 = blockIdx.x * 16;
  const int ll = lane & 15, hi = lane >> 4;
  const int n0 = wv * 32;

  // ---- stage block-shared activations via async global->LDS ----
  {
    const char* gx = (const char*)(Xt + (size_t)row0 * HS);   // contiguous 16KB
    char* lx = (char*)&s_x[0][0];
    for (int c4 = tid; c4 < 1024; c4 += 256)
      async_cp16(lx + c4 * 16, gx + c4 * 16);
    if (!firstStep) {
      const char* g0 = (const char*)(h0prevG + (size_t)row0 * SF);  // 8KB
      const char* g1 = (const char*)(h1prevG + (size_t)row0 * SF);  // 8KB
      char* l0 = (char*)&s_h0p[0][0];
      char* l1 = (char*)&s_h1p[0][0];
      for (int c4 = tid; c4 < 512; c4 += 256) {
        async_cp16(l0 + c4 * 16, g0 + c4 * 16);
        async_cp16(l1 + c4 * 16, g1 + c4 * 16);
      }
    }
    async_wait0();
    __syncthreads();
  }

  const v8f vzero = {0,0,0,0,0,0,0,0};
  const bool zrow = firstStep || (dones[row0 + ll] != 0);
  v8f acc[4][2]; BFrag a, b;

  // ================= layer 0 =================
#pragma unroll
  for (int g = 0; g < 4; ++g)
#pragma unroll
    for (int s = 0; s < 2; ++s) acc[g][s] = vzero;

  for (int k0 = 0; k0 < HS; k0 += 32) {
    load_fragA_bf16(a, &s_x[0][0], 512, 0, k0, lane);
#pragma unroll
    for (int g = 0; g < 4; ++g)
#pragma unroll
      for (int s = 0; s < 2; ++s) {
        load_fragB(b, Wih0B, HS, g * SF + n0 + s * 16, k0, lane);
        acc[g][s] = WMMA_BF16(a, b, acc[g][s]);
      }
  }
  for (int k0 = 0; k0 < SF; k0 += 32) {
    if (zrow) zero_frag(a);
    else      load_fragA_bf16(a, &s_h0p[0][0], SF, 0, k0, lane);
#pragma unroll
    for (int g = 0; g < 4; ++g)
#pragma unroll
      for (int s = 0; s < 2; ++s) {
        load_fragB(b, Whh0B, SF, g * SF + n0 + s * 16, k0, lane);
        acc[g][s] = WMMA_BF16(a, b, acc[g][s]);
      }
  }
#pragma unroll
  for (int s = 0; s < 2; ++s) {
    const int col = n0 + s * 16 + ll;
    const float bI = bih0[0 * SF + col] + bhh0[0 * SF + col];
    const float bF = bih0[1 * SF + col] + bhh0[1 * SF + col];
    const float bG = bih0[2 * SF + col] + bhh0[2 * SF + col];
    const float bO = bih0[3 * SF + col] + bhh0[3 * SF + col];
#pragma unroll
    for (int v = 0; v < 8; ++v) {
      const int row = row0 + hi * 8 + v;
      const bool zr = firstStep || (dones[row] != 0);
      const float iv = acc[0][s][v] + bI;
      const float fv = acc[1][s][v] + bF;
      const float gv = acc[2][s][v] + bG;
      const float ov = acc[3][s][v] + bO;
      const float cp = zr ? 0.f : c0[(size_t)row * SF + col];
      const float c2 = sigmoidf_(fv) * cp + sigmoidf_(iv) * tanhf(gv);
      const float h2 = sigmoidf_(ov) * tanhf(c2);
      const unsigned short hb = f2bf(h2);
      c0[(size_t)row * SF + col] = c2;
      s_h0[hi * 8 + v][col]      = hb;
      h0curG[(size_t)row * SF + col] = hb;
      if (lastStep) h0F[(size_t)row * SF + col] = h2;
    }
  }
  __syncthreads();

  // ================= layer 1 =================
#pragma unroll
  for (int g = 0; g < 4; ++g)
#pragma unroll
    for (int s = 0; s < 2; ++s) acc[g][s] = vzero;

  for (int k0 = 0; k0 < SF; k0 += 32) {
    load_fragA_bf16(a, &s_h0[0][0], SF, 0, k0, lane);
#pragma unroll
    for (int g = 0; g < 4; ++g)
#pragma unroll
      for (int s = 0; s < 2; ++s) {
        load_fragB(b, Wih1B, SF, g * SF + n0 + s * 16, k0, lane);
        acc[g][s] = WMMA_BF16(a, b, acc[g][s]);
      }
  }
  for (int k0 = 0; k0 < SF; k0 += 32) {
    if (zrow) zero_frag(a);
    else      load_fragA_bf16(a, &s_h1p[0][0], SF, 0, k0, lane);
#pragma unroll
    for (int g = 0; g < 4; ++g)
#pragma unroll
      for (int s = 0; s < 2; ++s) {
        load_fragB(b, Whh1B, SF, g * SF + n0 + s * 16, k0, lane);
        acc[g][s] = WMMA_BF16(a, b, acc[g][s]);
      }
  }
#pragma unroll
  for (int s = 0; s < 2; ++s) {
    const int col = n0 + s * 16 + ll;
    const float bI = bih1[0 * SF + col] + bhh1[0 * SF + col];
    const float bF = bih1[1 * SF + col] + bhh1[1 * SF + col];
    const float bG = bih1[2 * SF + col] + bhh1[2 * SF + col];
    const float bO = bih1[3 * SF + col] + bhh1[3 * SF + col];
#pragma unroll
    for (int v = 0; v < 8; ++v) {
      const int row = row0 + hi * 8 + v;
      const bool zr = firstStep || (dones[row] != 0);
      const float iv = acc[0][s][v] + bI;
      const float fv = acc[1][s][v] + bF;
      const float gv = acc[2][s][v] + bG;
      const float ov = acc[3][s][v] + bO;
      const float cp = zr ? 0.f : c1[(size_t)row * SF + col];
      const float c2 = sigmoidf_(fv) * cp + sigmoidf_(iv) * tanhf(gv);
      const float h2 = sigmoidf_(ov) * tanhf(c2);
      c1[(size_t)row * SF + col]     = c2;
      h1outF[(size_t)row * SF + col] = h2;
      h1outB[(size_t)row * SF + col] = f2bf(h2);
    }
  }
}

// Fused head: forward model, inverse model (mu/std), losses, intrinsic reward.
// One wave handles 16 flattened (t,b) rows; 2 waves per block.
__global__ void __launch_bounds__(64)
k_head(const unsigned short* __restrict__ outsB,   // (T1,B,SF) bf16
       const float* __restrict__ outsF,            // (T1,B,SF) f32
       const float* __restrict__ actF,             // (T,B,AD) f32
       const unsigned short* __restrict__ Wf1B, const float* __restrict__ bf1,
       const unsigned short* __restrict__ Wf2B, const float* __restrict__ bf2,
       const unsigned short* __restrict__ Wm1B, const float* __restrict__ bm1,
       const unsigned short* __restrict__ Wm2B, const float* __restrict__ bm2,
       const unsigned short* __restrict__ Ws1B, const float* __restrict__ bs1,
       const unsigned short* __restrict__ Ws2B, const float* __restrict__ bs2,
       float* __restrict__ out) {
  __shared__ unsigned short s_hid[2][16][512];   // 32KB
  __shared__ unsigned short s_pred[2][16][256];  // 16KB
  __shared__ float          s_mu[2][16][32];     // 4KB

  const int lane = threadIdx.x & 31;
  const int wv   = threadIdx.x >> 5;
  const int ll = lane & 15, hi = lane >> 4;
  const int gr0 = (blockIdx.x * 2 + wv) * 16;    // flattened (t,b), tile-aligned
  const int t  = gr0 / B;
  const int b0 = gr0 % B;

  const unsigned short* sfB = outsB + (size_t)t * B * SF + (size_t)b0 * SF;
  const float* act = actF + (size_t)t * B * AD + (size_t)b0 * AD;
  const float* nsf = outsF + (size_t)(t + 1) * B * SF + (size_t)b0 * SF;

  const v8f vzero = {0,0,0,0,0,0,0,0};
  BFrag a, b;
  float fsum = 0.f, lpsum = 0.f;

  // ---- hidF = relu([sf, act] @ Wf1^T + bf1) -> LDS (bf16) ----
  for (int nt = 0; nt < 32; ++nt) {
    v8f acc = vzero;
    for (int ck = 0; ck < 9; ++ck) {
      const int k0 = ck * 32;
      if (k0 < SF) load_fragA_bf16(a, sfB, SF, 0, k0, lane);
      else         load_fragA_f32(a, act, AD, 0, k0 - SF, lane);
      load_fragB(b, Wf1B, SF + AD, nt * 16, k0, lane);
      acc = WMMA_BF16(a, b, acc);
    }
    const int col = nt * 16 + ll;
    const float bb = bf1[col];
#pragma unroll
    for (int v = 0; v < 8; ++v) {
      float x = acc[v] + bb;
      x = x > 0.f ? x : 0.f;
      s_hid[wv][hi * 8 + v][col] = f2bf(x);
    }
  }
  __syncthreads();

  // ---- pred = hidF @ Wf2^T + bf2 ; fl = (pred-nsf)^2 ----
  float flRow[8];
#pragma unroll
  for (int v = 0; v < 8; ++v) flRow[v] = 0.f;
  for (int nt = 0; nt < 16; ++nt) {
    v8f acc = vzero;
    for (int ck = 0; ck < 16; ++ck) {
      load_fragA_bf16(a, &s_hid[wv][0][0], 512, 0, ck * 32, lane);
      load_fragB(b, Wf2B, HS, nt * 16, ck * 32, lane);
      acc = WMMA_BF16(a, b, acc);
    }
    const int col = nt * 16 + ll;
    const float bb = bf2[col];
#pragma unroll
    for (int v = 0; v < 8; ++v) {
      const int r = hi * 8 + v;
      const float p = acc[v] + bb;
      const float d = p - nsf[(size_t)r * SF + col];
      const float fl = d * d;
      fsum += fl;
      flRow[v] += fl;
      s_pred[wv][r][col] = f2bf(p);
    }
  }
  // intrinsic_reward[t, b]: reduce each row across the 16 column-lanes
#pragma unroll
  for (int v = 0; v < 8; ++v) {
    float vs = flRow[v];
#pragma unroll
    for (int m = 8; m >= 1; m >>= 1) vs += __shfl_xor(vs, m, 32);
    if (ll == 0) out[2 + (size_t)t * B + b0 + hi * 8 + v] = vs;
  }
  __syncthreads();

  // ---- hidM = relu([sf, pred] @ Wm1^T + bm1) -> LDS ----
  for (int nt = 0; nt < 32; ++nt) {
    v8f acc = vzero;
    for (int ck = 0; ck < 16; ++ck) {
      const int k0 = ck * 32;
      if (k0 < SF) load_fragA_bf16(a, sfB, SF, 0, k0, lane);
      else         load_fragA_bf16(a, &s_pred[wv][0][0], SF, 0, k0 - SF, lane);
      load_fragB(b, Wm1B, 2 * SF, nt * 16, k0, lane);
      acc = WMMA_BF16(a, b, acc);
    }
    const int col = nt * 16 + ll;
    const float bb = bm1[col];
#pragma unroll
    for (int v = 0; v < 8; ++v) {
      float x = acc[v] + bb;
      x = x > 0.f ? x : 0.f;
      s_hid[wv][hi * 8 + v][col] = f2bf(x);
    }
  }
  __syncthreads();

  // ---- mu = tanh(hidM @ Wm2^T + bm2) -> LDS (f32) ----
  for (int nt = 0; nt < 2; ++nt) {
    v8f acc = vzero;
    for (int ck = 0; ck < 16; ++ck) {
      load_fragA_bf16(a, &s_hid[wv][0][0], 512, 0, ck * 32, lane);
      load_fragB(b, Wm2B, HS, nt * 16, ck * 32, lane);
      acc = WMMA_BF16(a, b, acc);
    }
    const int col = nt * 16 + ll;
    const float bb = bm2[col];
#pragma unroll
    for (int v = 0; v < 8; ++v)
      s_mu[wv][hi * 8 + v][col] = tanhf(acc[v] + bb);
  }
  __syncthreads();

  // ---- hidS = relu([sf, pred] @ Ws1^T + bs1) -> LDS (overwrites hidM) ----
  for (int nt = 0; nt < 32; ++nt) {
    v8f acc = vzero;
    for (int ck = 0; ck < 16; ++ck) {
      const int k0 = ck * 32;
      if (k0 < SF) load_fragA_bf16(a, sfB, SF, 0, k0, lane);
      else         load_fragA_bf16(a, &s_pred[wv][0][0], SF, 0, k0 - SF, lane);
      load_fragB(b, Ws1B, 2 * SF, nt * 16, k0, lane);
      acc = WMMA_BF16(a, b, acc);
    }
    const int col = nt * 16 + ll;
    const float bb = bs1[col];
#pragma unroll
    for (int v = 0; v < 8; ++v) {
      float x = acc[v] + bb;
      x = x > 0.f ? x : 0.f;
      s_hid[wv][hi * 8 + v][col] = f2bf(x);
    }
  }
  __syncthreads();

  // ---- std = softplus(hidS @ Ws2^T + bs2); log-prob ----
  for (int nt = 0; nt < 2; ++nt) {
    v8f acc = vzero;
    for (int ck = 0; ck < 16; ++ck) {
      load_fragA_bf16(a, &s_hid[wv][0][0], 512, 0, ck * 32, lane);
      load_fragB(b, Ws2B, HS, nt * 16, ck * 32, lane);
      acc = WMMA_BF16(a, b, acc);
    }
    const int col = nt * 16 + ll;
    const float bb = bs2[col];
#pragma unroll
    for (int v = 0; v < 8; ++v) {
      const int r = hi * 8 + v;
      const float x  = acc[v] + bb;
      const float sd = (x > 20.f) ? x : log1pf(expf(x));
      const float av = act[(size_t)r * AD + col];
      const float mu = s_mu[wv][r][col];
      const float z  = (av - mu) / sd;
      lpsum += -0.5f * z * z - logf(sd) - 0.918938533204672742f;
    }
  }

  // ---- loss reductions ----
#pragma unroll
  for (int m = 16; m >= 1; m >>= 1) {
    fsum  += __shfl_xor(fsum, m, 32);
    lpsum += __shfl_xor(lpsum, m, 32);
  }
  if (lane == 0) {
    atomicAdd(&out[0], fsum  * (1.0f / 33554432.0f));   // mean over T*B*SF
    atomicAdd(&out[1], -lpsum * (1.0f / 4194304.0f));   // -mean over T*B*AD
  }
}

__global__ void __launch_bounds__(256)
k_hidden(const float* __restrict__ h0F, const float* __restrict__ h1F,
         const float* __restrict__ c0, const float* __restrict__ c1,
         float* __restrict__ outH) {
  int i = blockIdx.x * blockDim.x + threadIdx.x;
  if (i < B * SF) {
    outH[i]              = h0F[i];
    outH[B * SF + i]     = h1F[i];
    outH[2 * B * SF + i] = c0[i];
    outH[3 * B * SF + i] = c1[i];
  }
}

// ---------------- host launch ----------------
extern "C" void kernel_launch(void* const* d_in, const int* in_sizes, int n_in,
                              void* d_out, int out_size, void* d_ws, size_t ws_size,
                              hipStream_t stream) {
  (void)in_sizes; (void)n_in; (void)out_size; (void)ws_size;
  const float* states = (const float*)d_in[0];
  const float* action = (const float*)d_in[1];
  const int*   dones  = (const int*)d_in[2];
  const float* Wfe  = (const float*)d_in[3];
  const float* bfe  = (const float*)d_in[4];
  const float* Wih0 = (const float*)d_in[5];
  const float* Whh0 = (const float*)d_in[6];
  const float* bih0 = (const float*)d_in[7];
  const float* bhh0 = (const float*)d_in[8];
  const float* Wih1 = (const float*)d_in[9];
  const float* Whh1 = (const float*)d_in[10];
  const float* bih1 = (const float*)d_in[11];
  const float* bhh1 = (const float*)d_in[12];
  const float* Wf1  = (const float*)d_in[13];
  const float* bf1  = (const float*)d_in[14];
  const float* Wf2  = (const float*)d_in[15];
  const float* bf2  = (const float*)d_in[16];
  const float* Wm1  = (const float*)d_in[17];
  const float* bm1  = (const float*)d_in[18];
  const float* Wm2  = (const float*)d_in[19];
  const float* bm2  = (const float*)d_in[20];
  const float* Ws1  = (const float*)d_in[21];
  const float* bs1  = (const float*)d_in[22];
  const float* Ws2  = (const float*)d_in[23];
  const float* bs2  = (const float*)d_in[24];
  float* out = (float*)d_out;

  typedef unsigned short us;
  char* p = (char*)d_ws;
  auto carve = [&](size_t elems, size_t esize) -> char* {
    char* r = p;
    p += (elems * esize + 255) & ~(size_t)255;
    return r;
  };
  us* WfeB   = (us*)carve((size_t)HS * SD, 2);
  us* Wih0B  = (us*)carve((size_t)4 * SF * HS, 2);
  us* Whh0B  = (us*)carve((size_t)4 * SF * SF, 2);
  us* Wih1B  = (us*)carve((size_t)4 * SF * SF, 2);
  us* Whh1B  = (us*)carve((size_t)4 * SF * SF, 2);
  us* Wf1B   = (us*)carve((size_t)HS * (SF + AD), 2);
  us* Wf2B   = (us*)carve((size_t)SF * HS, 2);
  us* Wm1B   = (us*)carve((size_t)HS * 2 * SF, 2);
  us* Wm2B   = (us*)carve((size_t)AD * HS, 2);
  us* Ws1B   = (us*)carve((size_t)HS * 2 * SF, 2);
  us* Ws2B   = (us*)carve((size_t)AD * HS, 2);
  us* featsB = (us*)carve((size_t)T1 * B * HS, 2);
  us* outsB  = (us*)carve((size_t)T1 * B * SF, 2);
  float* outsF = (float*)carve((size_t)T1 * B * SF, 4);
  us* h0Ba   = (us*)carve((size_t)B * SF, 2);
  us* h0Bb   = (us*)carve((size_t)B * SF, 2);
  float* h0F = (float*)carve((size_t)B * SF, 4);
  float* c0  = (float*)carve((size_t)B * SF, 4);
  float* c1  = (float*)carve((size_t)B * SF, 4);

  // weight conversions f32 -> bf16
  struct CV { const float* s; us* d; int n; } cv[] = {
    {Wfe, WfeB, HS * SD}, {Wih0, Wih0B, 4 * SF * HS}, {Whh0, Whh0B, 4 * SF * SF},
    {Wih1, Wih1B, 4 * SF * SF}, {Whh1, Whh1B, 4 * SF * SF},
    {Wf1, Wf1B, HS * (SF + AD)}, {Wf2, Wf2B, SF * HS},
    {Wm1, Wm1B, HS * 2 * SF}, {Wm2, Wm2B, AD * HS},
    {Ws1, Ws1B, HS * 2 * SF}, {Ws2, Ws2B, AD * HS},
  };
  for (int i = 0; i < (int)(sizeof(cv) / sizeof(cv[0])); ++i)
    k_f32_to_bf16<<<dim3((cv[i].n + 255) / 256), dim3(256), 0, stream>>>(cv[i].s, cv[i].d, cv[i].n);

  k_init<<<dim3((B * SF + 255) / 256), dim3(256), 0, stream>>>(out, c0, c1);

  // feature extractor GEMM
  k_feats<<<dim3((T1 * B) / 64, HS / 64), dim3(128), 0, stream>>>(states, WfeB, bfe, featsB);

  // sequential LSTM scan: one fused kernel per timestep
  for (int t = 0; t < T1; ++t) {
    const us* hprev0 = (t & 1) ? h0Bb : h0Ba;
    us*       hcur0  = (t & 1) ? h0Ba : h0Bb;
    const us* hprev1 = (t == 0) ? outsB : outsB + (size_t)(t - 1) * B * SF;
    k_lstm_step<<<dim3(B / 16), dim3(256), 0, stream>>>(
        featsB + (size_t)t * B * HS,
        hprev0, hcur0, h0F,
        hprev1, outsB + (size_t)t * B * SF, outsF + (size_t)t * B * SF,
        Wih0B, Whh0B, bih0, bhh0,
        Wih1B, Whh1B, bih1, bhh1,
        dones + (size_t)t * B, c0, c1,
        t == 0 ? 1 : 0, t == T1 - 1 ? 1 : 0);
  }

  // fused heads (forward model + inverse model + losses)
  k_head<<<dim3((T * B) / 32), dim3(64), 0, stream>>>(
      outsB, outsF, action,
      Wf1B, bf1, Wf2B, bf2, Wm1B, bm1, Wm2B, bm2, Ws1B, bs1, Ws2B, bs2, out);

  // hidden = concat([hT, cT]) -> (4, B, SF)
  k_hidden<<<dim3((B * SF + 255) / 256), dim3(256), 0, stream>>>(
      h0F, outsF + (size_t)(T1 - 1) * B * SF, c0, c1,
      out + 2 + (size_t)T * B);
}
